// IncidenceBiasAttention_15109694947391
// MI455X (gfx1250) — compile-verified, matrix operand order checked by reading, not run on hardware
//
#include <hip/hip_runtime.h>
#include <hip/hip_bf16.h>
#include <cmath>

typedef __bf16 bf16_t;
typedef __attribute__((ext_vector_type(16))) __bf16 v16bf;
typedef __attribute__((ext_vector_type(8)))  __bf16 v8bf;
typedef __attribute__((ext_vector_type(8)))  float  v8f;
typedef __attribute__((ext_vector_type(2)))  float  v2f;

#define NEG_BIG (-1.0e30f)

// Build a 16-element bf16 A/B fragment from two contiguous 8x-bf16 (16B) loads.
// Per CDNA5 WMMA layout, a lane's 16 K-values are two contiguous runs of 8.
__device__ __forceinline__ v16bf load_frag16(const bf16_t* p) {
  v8bf lo = *(const v8bf*)(p);
  v8bf hi = *(const v8bf*)(p + 16);
  return __builtin_shufflevector(lo, hi, 0,1,2,3,4,5,6,7,8,9,10,11,12,13,14,15);
}

__device__ __forceinline__ float red_max16(float v) {
  #pragma unroll
  for (int m = 1; m < 16; m <<= 1) v = fmaxf(v, __shfl_xor(v, m, 32));
  return v;
}
__device__ __forceinline__ float red_sum16(float v) {
  #pragma unroll
  for (int m = 1; m < 16; m <<= 1) v += __shfl_xor(v, m, 32);
  return v;
}

// ---------------------------------------------------------------------------
// f32 -> bf16 conversion
// ---------------------------------------------------------------------------
__global__ void cvt_f32_bf16(const float* __restrict__ src, bf16_t* __restrict__ dst, int n) {
  int i = blockIdx.x * blockDim.x + threadIdx.x;
  if (i < n) dst[i] = (bf16_t)src[i];
}

// ---------------------------------------------------------------------------
// C[M,N] (f32) = A[M,K](bf16) * B[N,K](bf16)^T  (+ bias[N])
// One wave -> one 32x64 C tile (2x4 grid of 16x16 WMMA tiles).
// Per K-step of 32: 2 A frags + 4 B frags feed 8 v_wmma_f32_16x16x32_bf16,
// giving 2.7x register-level reuse vs 1 tile/wave.
// Requires M % 32 == 0, N % 64 == 0 (true for all shapes here).
// ---------------------------------------------------------------------------
__global__ void gemm_bf16_wmma(const bf16_t* __restrict__ A, const bf16_t* __restrict__ B,
                               const float* __restrict__ bias, int has_bias,
                               float* __restrict__ C, int M, int N, int K, int total_waves) {
  int wave = blockIdx.x * (blockDim.x >> 5) + (threadIdx.x >> 5);
  if (wave >= total_waves) return;
  int lane  = threadIdx.x & 31;
  int r     = lane & 15;
  int khalf = (lane < 16) ? 0 : 8;

  int ntiles = N >> 6;                 // 64-wide column tiles
  int tm = wave / ntiles;
  int tn = wave - tm * ntiles;

  const bf16_t* arow0 = A + (size_t)(tm * 32 + r) * K;
  const bf16_t* arow1 = arow0 + (size_t)16 * K;
  const bf16_t* brow0 = B + (size_t)(tn * 64 + r) * K;
  const bf16_t* brow1 = brow0 + (size_t)16 * K;
  const bf16_t* brow2 = brow0 + (size_t)32 * K;
  const bf16_t* brow3 = brow0 + (size_t)48 * K;

  v8f acc[8];
  #pragma unroll
  for (int i = 0; i < 8; ++i) acc[i] = (v8f){0.f,0.f,0.f,0.f,0.f,0.f,0.f,0.f};

  for (int k = 0; k < K; k += 32) {
    v16bf a0 = load_frag16(arow0 + k + khalf);
    v16bf a1 = load_frag16(arow1 + k + khalf);
    v16bf b0 = load_frag16(brow0 + k + khalf);
    v16bf b1 = load_frag16(brow1 + k + khalf);
    v16bf b2 = load_frag16(brow2 + k + khalf);
    v16bf b3 = load_frag16(brow3 + k + khalf);
    acc[0] = __builtin_amdgcn_wmma_f32_16x16x32_bf16(false, a0, false, b0, (short)0, acc[0], false, false);
    acc[1] = __builtin_amdgcn_wmma_f32_16x16x32_bf16(false, a0, false, b1, (short)0, acc[1], false, false);
    acc[2] = __builtin_amdgcn_wmma_f32_16x16x32_bf16(false, a0, false, b2, (short)0, acc[2], false, false);
    acc[3] = __builtin_amdgcn_wmma_f32_16x16x32_bf16(false, a0, false, b3, (short)0, acc[3], false, false);
    acc[4] = __builtin_amdgcn_wmma_f32_16x16x32_bf16(false, a1, false, b0, (short)0, acc[4], false, false);
    acc[5] = __builtin_amdgcn_wmma_f32_16x16x32_bf16(false, a1, false, b1, (short)0, acc[5], false, false);
    acc[6] = __builtin_amdgcn_wmma_f32_16x16x32_bf16(false, a1, false, b2, (short)0, acc[6], false, false);
    acc[7] = __builtin_amdgcn_wmma_f32_16x16x32_bf16(false, a1, false, b3, (short)0, acc[7], false, false);
  }

  int qoff = (lane < 16) ? 0 : 8;
  #pragma unroll
  for (int ai = 0; ai < 2; ++ai) {
    int rbase = tm * 32 + ai * 16 + qoff;
    #pragma unroll
    for (int bi = 0; bi < 4; ++bi) {
      int col  = tn * 64 + bi * 16 + r;
      float bv = has_bias ? bias[col] : 0.f;
      v8f a = acc[ai * 4 + bi];
      #pragma unroll
      for (int i = 0; i < 8; ++i)
        C[(size_t)(rbase + i) * N + col] = a[i] + bv;
    }
  }
}

// ---------------------------------------------------------------------------
// qkv f32 [B*T, 3*D] -> Qb/Kb [B,H,T,64] bf16, Vt [B,H,64,T] bf16 (transposed)
// ---------------------------------------------------------------------------
__global__ void prep_attn(const float* __restrict__ qkv,
                          bf16_t* __restrict__ Qb, bf16_t* __restrict__ Kb, bf16_t* __restrict__ Vt,
                          int B, int T, int H) {
  const int dh = 64;
  const int D  = H * dh;
  int idx = blockIdx.x * blockDim.x + threadIdx.x;
  int total = B * T * H * dh;
  if (idx >= total) return;
  int d  = idx % dh;
  int h  = (idx / dh) % H;
  int bt = idx / (dh * H);
  int t  = bt % T;
  int b  = bt / T;

  size_t rowb = (size_t)bt * (3 * D);
  float q = qkv[rowb + 0 * D + h * dh + d];
  float k = qkv[rowb + 1 * D + h * dh + d];
  float v = qkv[rowb + 2 * D + h * dh + d];

  size_t o = ((size_t)(b * H + h) * T + t) * dh + d;
  Qb[o] = (bf16_t)q;
  Kb[o] = (bf16_t)k;
  Vt[((size_t)(b * H + h) * dh + d) * T + t] = (bf16_t)v;
}

// ---------------------------------------------------------------------------
// lines f32 [B*T, 4*PH]  (blocks: w1|w2|r1|r2, each PH=pd*H cols)
//   -> R8 [B,H,T,8] f32 (normalized read lines, padded)
//   -> W8 [B,H,T,8] f32 (J6-applied normalized write lines, padded)
// w1 uses row t-1 (x_prev); t==0 -> zeros.
// ---------------------------------------------------------------------------
__device__ __forceinline__ void exterior6(const float a[4], const float b[4], float L[6]) {
  L[0] = a[0]*b[1] - a[1]*b[0];
  L[1] = a[0]*b[2] - a[2]*b[0];
  L[2] = a[0]*b[3] - a[3]*b[0];
  L[3] = a[1]*b[2] - a[2]*b[1];
  L[4] = a[1]*b[3] - a[3]*b[1];
  L[5] = a[2]*b[3] - a[3]*b[2];
  float n2 = L[0]*L[0]+L[1]*L[1]+L[2]*L[2]+L[3]*L[3]+L[4]*L[4]+L[5]*L[5];
  float inv = 1.f / fmaxf(sqrtf(n2), 1e-12f);
  #pragma unroll
  for (int j = 0; j < 6; ++j) L[j] *= inv;
}

__global__ void prep_lines(const float* __restrict__ lines,
                           float* __restrict__ R8, float* __restrict__ W8,
                           int B, int T, int H, int LW) {
  int idx = blockIdx.x * blockDim.x + threadIdx.x;
  int total = B * T * H;
  if (idx >= total) return;
  int h  = idx % H;
  int bt = idx / H;
  int t  = bt % T;
  int b  = bt / T;
  int PH = LW >> 2;
  int pd = PH / H;

  const float* row = lines + (size_t)bt * LW;
  float w1[4] = {0.f, 0.f, 0.f, 0.f}, w2[4], r1[4], r2[4];
  if (t > 0) {
    const float* prow = lines + (size_t)(bt - 1) * LW;
    #pragma unroll
    for (int c = 0; c < 4; ++c) w1[c] = prow[0 * PH + h * pd + c];
  }
  #pragma unroll
  for (int c = 0; c < 4; ++c) {
    w2[c] = row[1 * PH + h * pd + c];
    r1[c] = row[2 * PH + h * pd + c];
    r2[c] = row[3 * PH + h * pd + c];
  }
  float Lw[6], Lr[6];
  exterior6(w1, w2, Lw);
  exterior6(r1, r2, Lr);

  size_t base = ((size_t)(b * H + h) * T + t) * 8;
  #pragma unroll
  for (int j = 0; j < 6; ++j) R8[base + j] = Lr[j];
  R8[base + 6] = 0.f; R8[base + 7] = 0.f;
  // (L @ J6): [L5, -L4, L3, L2, -L1, L0]
  W8[base + 0] =  Lw[5];
  W8[base + 1] = -Lw[4];
  W8[base + 2] =  Lw[3];
  W8[base + 3] =  Lw[2];
  W8[base + 4] = -Lw[1];
  W8[base + 5] =  Lw[0];
  W8[base + 6] = 0.f; W8[base + 7] = 0.f;
}

// ---------------------------------------------------------------------------
// Flash attention with incidence bias.
// One wave: 16 Q rows of one (b,h). Keys stepped 32 at a time (causal).
//   S    = Q K^T           : 2x v_wmma_f32_16x16x32_bf16 per 16-col half
//   bias = R8 . W8 (6-dim) : 2x v_wmma_f32_16x16x4_f32  per 16-col half
//   PV   = P V             : 4x v_wmma_f32_16x16x32_bf16 (dh = 4 x 16)
// P redistributed C-layout -> A-layout via LDS.
// ---------------------------------------------------------------------------
#define LDS_STRIDE 40   // 32 cols + pad; keeps 16B alignment (80B row pitch)

__global__ void flash_attn_wmma(const bf16_t* __restrict__ Qb, const bf16_t* __restrict__ Kb,
                                const bf16_t* __restrict__ Vt,
                                const float* __restrict__ R8, const float* __restrict__ W8,
                                const float* __restrict__ decay_logits,
                                const float* __restrict__ bias_scale,
                                bf16_t* __restrict__ attn_out,
                                int B, int T, int H) {
  __shared__ __align__(16) bf16_t lds_p[8 * 16 * LDS_STRIDE];
  const int dh = 64;
  const int D  = H * dh;

  int wave   = blockIdx.x * (blockDim.x >> 5) + (threadIdx.x >> 5);
  int qt_per = T >> 4;
  int total  = B * H * qt_per;
  if (wave >= total) return;

  int lane  = threadIdx.x & 31;
  int r     = lane & 15;
  int khalf = (lane < 16) ? 0 : 8;   // bf16 K-run start within 32-chunk
  int fhalf = (lane < 16) ? 0 : 2;   // f32  K-run start within 4-chunk
  int qoff  = (lane < 16) ? 0 : 8;   // C-layout row offset

  int qt = wave % qt_per;
  int h  = (wave / qt_per) % H;
  int b  = wave / (qt_per * H);
  int q0 = qt << 4;

  const bf16_t* Qbh = Qb + (size_t)(b * H + h) * T * dh;
  const bf16_t* Kbh = Kb + (size_t)(b * H + h) * T * dh;
  const bf16_t* Vth = Vt + (size_t)(b * H + h) * dh * T;
  const float*  R8h = R8 + (size_t)(b * H + h) * T * 8;
  const float*  W8h = W8 + (size_t)(b * H + h) * T * 8;

  float dl    = decay_logits[h];
  float decay = 1.f / (1.f + __expf(-dl));
  float l2d   = log2f(decay);
  float bs    = bias_scale[h];
  const float scale = 0.125f;  // dh^-0.5, dh = 64

  // Loop-invariant A fragments: Q (bf16, dh=64 -> 2 chunks) and read-lines (f32)
  const bf16_t* qrow = Qbh + (size_t)(q0 + r) * dh;
  v16bf qa0 = load_frag16(qrow + khalf);
  v16bf qa1 = load_frag16(qrow + 32 + khalf);
  const float* rrow = R8h + (size_t)(q0 + r) * 8;
  v2f ra0 = *(const v2f*)(rrow + fhalf);
  v2f ra1 = *(const v2f*)(rrow + 4 + fhalf);

  float m_i[8], L_i[8];
  v8f o0 = {0.f,0.f,0.f,0.f,0.f,0.f,0.f,0.f};
  v8f o1 = o0, o2 = o0, o3 = o0;
  #pragma unroll
  for (int i = 0; i < 8; ++i) { m_i[i] = NEG_BIG; L_i[i] = 0.f; }

  bf16_t* lp = lds_p + (threadIdx.x >> 5) * (16 * LDS_STRIDE);
  const v8f vz = {0.f,0.f,0.f,0.f,0.f,0.f,0.f,0.f};
  int qmax = q0 + 15;

  for (int k0 = 0; k0 <= qmax; k0 += 32) {
    // ---- scores + bias for the two 16-key halves ----
    v8f s0, s1, bb0, bb1;
    {
      const bf16_t* krow = Kbh + (size_t)(k0 + r) * dh;
      v16bf kb0 = load_frag16(krow + khalf);
      v16bf kb1 = load_frag16(krow + 32 + khalf);
      v8f acc = vz;
      acc = __builtin_amdgcn_wmma_f32_16x16x32_bf16(false, qa0, false, kb0, (short)0, acc, false, false);
      acc = __builtin_amdgcn_wmma_f32_16x16x32_bf16(false, qa1, false, kb1, (short)0, acc, false, false);
      s0 = acc;
      const float* wrow = W8h + (size_t)(k0 + r) * 8;
      v2f wb0 = *(const v2f*)(wrow + fhalf);
      v2f wb1 = *(const v2f*)(wrow + 4 + fhalf);
      v8f bacc = vz;
      bacc = __builtin_amdgcn_wmma_f32_16x16x4_f32(false, ra0, false, wb0, (short)0, bacc, false, false);
      bacc = __builtin_amdgcn_wmma_f32_16x16x4_f32(false, ra1, false, wb1, (short)0, bacc, false, false);
      bb0 = bacc;
    }
    {
      const bf16_t* krow = Kbh + (size_t)(k0 + 16 + r) * dh;
      v16bf kb0 = load_frag16(krow + khalf);
      v16bf kb1 = load_frag16(krow + 32 + khalf);
      v8f acc = vz;
      acc = __builtin_amdgcn_wmma_f32_16x16x32_bf16(false, qa0, false, kb0, (short)0, acc, false, false);
      acc = __builtin_amdgcn_wmma_f32_16x16x32_bf16(false, qa1, false, kb1, (short)0, acc, false, false);
      s1 = acc;
      const float* wrow = W8h + (size_t)(k0 + 16 + r) * 8;
      v2f wb0 = *(const v2f*)(wrow + fhalf);
      v2f wb1 = *(const v2f*)(wrow + 4 + fhalf);
      v8f bacc = vz;
      bacc = __builtin_amdgcn_wmma_f32_16x16x4_f32(false, ra0, false, wb0, (short)0, bacc, false, false);
      bacc = __builtin_amdgcn_wmma_f32_16x16x4_f32(false, ra1, false, wb1, (short)0, bacc, false, false);
      bb1 = bacc;
    }

    // ---- logits, causal mask, decay weighting, online softmax ----
    #pragma unroll
    for (int i = 0; i < 8; ++i) {
      int q  = q0 + qoff + i;
      int d0 = q - (k0 + r);
      int d1 = q - (k0 + 16 + r);
      float w0 = (d0 > 0) ? bs * exp2f(l2d * (float)d0) : 0.f;
      float w1 = (d1 > 0) ? bs * exp2f(l2d * (float)d1) : 0.f;
      float l0 = (d0 < 0) ? NEG_BIG : (s0[i] * scale + w0 * bb0[i]);
      float l1 = (d1 < 0) ? NEG_BIG : (s1[i] * scale + w1 * bb1[i]);

      float tmax = red_max16(fmaxf(l0, l1));
      float mnew = fmaxf(m_i[i], tmax);
      float alpha = __expf(m_i[i] - mnew);
      m_i[i] = mnew;

      float p0 = __expf(l0 - mnew);
      float p1 = __expf(l1 - mnew);
      L_i[i] = L_i[i] * alpha + red_sum16(p0 + p1);

      int row = qoff + i;
      lp[row * LDS_STRIDE + r]      = (bf16_t)p0;
      lp[row * LDS_STRIDE + 16 + r] = (bf16_t)p1;

      o0[i] *= alpha; o1[i] *= alpha; o2[i] *= alpha; o3[i] *= alpha;
    }
    asm volatile("s_wait_dscnt 0" ::: "memory");

    // ---- P (A-layout from LDS) x V^T chunks ----
    v8bf plo = *(const v8bf*)(lp + r * LDS_STRIDE + khalf);
    v8bf phi = *(const v8bf*)(lp + r * LDS_STRIDE + khalf + 16);
    v16bf pa = __builtin_shufflevector(plo, phi, 0,1,2,3,4,5,6,7,8,9,10,11,12,13,14,15);

    v16bf vb;
    vb = load_frag16(Vth + (size_t)( 0 + r) * T + k0 + khalf);
    o0 = __builtin_amdgcn_wmma_f32_16x16x32_bf16(false, pa, false, vb, (short)0, o0, false, false);
    vb = load_frag16(Vth + (size_t)(16 + r) * T + k0 + khalf);
    o1 = __builtin_amdgcn_wmma_f32_16x16x32_bf16(false, pa, false, vb, (short)0, o1, false, false);
    vb = load_frag16(Vth + (size_t)(32 + r) * T + k0 + khalf);
    o2 = __builtin_amdgcn_wmma_f32_16x16x32_bf16(false, pa, false, vb, (short)0, o2, false, false);
    vb = load_frag16(Vth + (size_t)(48 + r) * T + k0 + khalf);
    o3 = __builtin_amdgcn_wmma_f32_16x16x32_bf16(false, pa, false, vb, (short)0, o3, false, false);
  }

  // ---- normalize, write bf16 attention output in [B,T,D] layout ----
  #pragma unroll
  for (int i = 0; i < 8; ++i) {
    float inv = 1.f / L_i[i];
    int t = q0 + qoff + i;
    size_t base = ((size_t)b * T + t) * D + h * dh;
    attn_out[base +  0 + r] = (bf16_t)(o0[i] * inv);
    attn_out[base + 16 + r] = (bf16_t)(o1[i] * inv);
    attn_out[base + 32 + r] = (bf16_t)(o2[i] * inv);
    attn_out[base + 48 + r] = (bf16_t)(o3[i] * inv);
  }
}

// ---------------------------------------------------------------------------
// Host-side orchestration
// ---------------------------------------------------------------------------
extern "C" void kernel_launch(void* const* d_in, const int* in_sizes, int n_in,
                              void* d_out, int out_size, void* d_ws, size_t ws_size,
                              hipStream_t stream) {
  const float* x      = (const float*)d_in[0];
  const float* w_qkv  = (const float*)d_in[1];
  const float* b_qkv  = (const float*)d_in[2];
  const float* w1w    = (const float*)d_in[3];
  const float* w2w    = (const float*)d_in[4];
  const float* w1r    = (const float*)d_in[5];
  const float* w2r    = (const float*)d_in[6];
  const float* w_out  = (const float*)d_in[7];
  const float* b_out  = (const float*)d_in[8];
  const float* dlog   = (const float*)d_in[9];
  const float* bscale = (const float*)d_in[10];

  const int B  = 2;
  const int D  = (int)(sqrt((double)in_sizes[7]) + 0.5);   // w_out is D x D
  const int H  = in_sizes[9];
  const int T  = in_sizes[0] / (B * D);
  const int PH = in_sizes[3] / D;    // pd*H rows per line weight
  const int LW = 4 * PH;             // concatenated line-proj output width
  const int M  = B * T;

  char* ws = (char*)d_ws;
  size_t off = 0;
  auto alloc = [&](size_t bytes) { size_t o = off; off += (bytes + 255) & ~(size_t)255; return o; };

  size_t o_xb    = alloc((size_t)M * D * 2);
  size_t o_wqkvb = alloc((size_t)3 * D * D * 2);
  size_t o_wlnb  = alloc((size_t)LW * D * 2);
  size_t o_woutb = alloc((size_t)D * D * 2);
  size_t o_qkv   = alloc((size_t)M * 3 * D * 4);
  size_t o_lines = alloc((size_t)M * LW * 4);
  size_t o_Qb    = alloc((size_t)M * D * 2);
  size_t o_Kb    = alloc((size_t)M * D * 2);
  size_t o_Vt    = alloc((size_t)M * D * 2);
  size_t o_R8    = alloc((size_t)B * H * T * 8 * 4);
  size_t o_W8    = alloc((size_t)B * H * T * 8 * 4);
  size_t o_attnb = alloc((size_t)M * D * 2);
  (void)ws_size;

  auto cvt = [&](const float* src, size_t dstoff, size_t n) {
    int nb = (int)((n + 255) / 256);
    cvt_f32_bf16<<<nb, 256, 0, stream>>>(src, (bf16_t*)(ws + dstoff), (int)n);
  };
  cvt(x,     o_xb,    (size_t)M * D);
  cvt(w_qkv, o_wqkvb, (size_t)3 * D * D);
  cvt(w1w,   o_wlnb + (size_t)0 * PH * D * 2, (size_t)PH * D);
  cvt(w2w,   o_wlnb + (size_t)1 * PH * D * 2, (size_t)PH * D);
  cvt(w1r,   o_wlnb + (size_t)2 * PH * D * 2, (size_t)PH * D);
  cvt(w2r,   o_wlnb + (size_t)3 * PH * D * 2, (size_t)PH * D);
  cvt(w_out, o_woutb, (size_t)D * D);

  auto gemm = [&](size_t aoff, size_t boff, const float* bias, int has_bias,
                  float* C, int m, int n, int k) {
    int waves = (m / 32) * (n / 64);
    int blocks = (waves + 7) / 8;
    gemm_bf16_wmma<<<blocks, 256, 0, stream>>>(
        (const bf16_t*)(ws + aoff), (const bf16_t*)(ws + boff),
        bias, has_bias, C, m, n, k, waves);
  };

  // qkv = x @ w_qkv^T + b_qkv
  gemm(o_xb, o_wqkvb, b_qkv, 1, (float*)(ws + o_qkv), M, 3 * D, D);
  // lines = x @ [w1w;w2w;w1r;w2r]^T
  gemm(o_xb, o_wlnb, nullptr, 0, (float*)(ws + o_lines), M, LW, D);

  {
    int total = M * D;
    prep_attn<<<(total + 255) / 256, 256, 0, stream>>>(
        (const float*)(ws + o_qkv), (bf16_t*)(ws + o_Qb), (bf16_t*)(ws + o_Kb),
        (bf16_t*)(ws + o_Vt), B, T, H);
  }
  {
    int total = M * H;
    prep_lines<<<(total + 255) / 256, 256, 0, stream>>>(
        (const float*)(ws + o_lines), (float*)(ws + o_R8), (float*)(ws + o_W8),
        B, T, H, LW);
  }
  {
    int waves = B * H * (T / 16);
    int blocks = (waves + 7) / 8;
    flash_attn_wmma<<<blocks, 256, 0, stream>>>(
        (const bf16_t*)(ws + o_Qb), (const bf16_t*)(ws + o_Kb), (const bf16_t*)(ws + o_Vt),
        (const float*)(ws + o_R8), (const float*)(ws + o_W8),
        dlog, bscale, (bf16_t*)(ws + o_attnb), B, T, H);
  }
  // out = attn @ w_out^T + b_out
  gemm(o_attnb, o_woutb, b_out, 1, (float*)d_out, M, D, D);
}
